// LocalWindowTransformerBlock_3229815406661
// MI455X (gfx1250) — compile-verified, matrix operand order checked by reading, not run on hardware
//
#include <hip/hip_runtime.h>
#include <math.h>

// ---------------------------------------------------------------------------
// Problem constants (B=16, H=W=64, C=384, HID=1536, HEADS=12, WS=8, SHIFT=4)
// ---------------------------------------------------------------------------
#define N_TOK   65536       // B*H*W tokens
#define C_DIM   384
#define HID_DIM 1536
#define QKV_N   1152
#define N_HEADS 12
#define BN_RS   0.9999950000374997f   // 1/sqrt(1+1e-5)

typedef __bf16 bf16_t;
typedef bf16_t v16bf __attribute__((ext_vector_type(16)));
typedef float  v8f   __attribute__((ext_vector_type(8)));

union FragU { v16bf v; uint4 q[2]; unsigned short s[16]; };

// Async global->LDS path (CDNA5). Probe-confirmed (round-2 diagnostic): arg0 is
// addrspace(1) pointer to a 16B int vector; LDS side is addrspace(3).
#if defined(__HIP_DEVICE_COMPILE__) && defined(__gfx1250__) && \
    __has_builtin(__builtin_amdgcn_global_load_async_to_lds_b128) && \
    __has_builtin(__builtin_amdgcn_s_wait_asynccnt)
#define USE_ASYNC_LDS 1
typedef int v4i_ __attribute__((vector_size(16)));
typedef __attribute__((address_space(1))) v4i_ g_v4i;
typedef __attribute__((address_space(3))) v4i_ l_v4i;
#else
#define USE_ASYNC_LDS 0
#endif

__device__ inline unsigned short f2bf_u(float f) {
  union { bf16_t h; unsigned short u; } c;
  c.h = (bf16_t)f;                       // hw v_cvt, RNE
  return c.u;
}

__device__ inline v8f wmma_bf16(FragU a, FragU b, v8f c) {
  return __builtin_amdgcn_wmma_f32_16x16x32_bf16(false, a.v, false, b.v,
                                                 (short)0, c, false, false);
}

__device__ inline float gelu_exact(float x) {
  return 0.5f * x * (1.0f + erff(x * 0.7071067811865476f));
}

// windowed-row m -> token-order source/destination index (roll -4 then window
// partition; the inverse scatter uses the same map by symmetry).
__device__ inline int swin_map(int m) {
  int b  = m >> 12;
  int r  = m & 4095;
  int w  = r >> 6, t = r & 63;
  int wh = w >> 3, ww = w & 7;
  int i  = t >> 3, j  = t & 7;
  int y  = (wh * 8 + i + 4) & 63;
  int x  = (ww * 8 + j + 4) & 63;
  return (b << 12) + (y << 6) + x;
}

__device__ inline int lab3(int v) { return v < 56 ? 0 : (v < 60 ? 1 : 2); }

// ---------------------------------------------------------------------------
// fp32 -> bf16 conversion
// ---------------------------------------------------------------------------
__global__ void f2bf_kernel(const float* __restrict__ in,
                            unsigned short* __restrict__ out, int n) {
  int i = blockIdx.x * 256 + threadIdx.x;
  if (i < n) out[i] = f2bf_u(in[i]);
}

// ---------------------------------------------------------------------------
// LayerNorm over C=384, one wave per token; optional swin gather on input row.
// ---------------------------------------------------------------------------
__global__ __launch_bounds__(256)
void ln_kernel(const float* __restrict__ in, const float* __restrict__ g,
               const float* __restrict__ b, unsigned short* __restrict__ out,
               int gather_swin) {
  int token = blockIdx.x * 8 + (threadIdx.x >> 5);
  int lane  = threadIdx.x & 31;
  int src   = gather_swin ? swin_map(token) : token;
  const float* row = in + (size_t)src * C_DIM;
  float vals[12], s = 0.f, s2 = 0.f;
  for (int i = 0; i < 12; i++) {
    float v = row[lane + i * 32];
    vals[i] = v; s += v; s2 += v * v;
  }
  for (int m = 16; m >= 1; m >>= 1) {
    s  += __shfl_xor(s,  m, 32);
    s2 += __shfl_xor(s2, m, 32);
  }
  float mean = s * (1.f / C_DIM);
  float var  = s2 * (1.f / C_DIM) - mean * mean;
  float rs   = rsqrtf(var + 1e-5f);
  unsigned short* orow = out + (size_t)token * C_DIM;
  for (int i = 0; i < 12; i++) {
    int c = lane + i * 32;
    orow[c] = f2bf_u((vals[i] - mean) * rs * g[c] + b[c]);
  }
}

// ---------------------------------------------------------------------------
// Generic bf16 WMMA GEMM: out[m,n] = sum_k A[m,k]*W[n,k] (+epilogue)
// Block tile 64x64, 4 waves, each wave 32x32 (2x2 wmma tiles), K-step 32.
// Double-buffered LDS; async global->LDS copies overlap stage i+1 with the
// WMMAs of stage i (one barrier per K-step).
// mode 0: out=f32 acc+bias
// mode 1: out[swin(m)] = acc+bias + res[swin(m)]          (proj + residual)
// mode 2: out = gelu(bn_g*BN_RS*(acc+bias)+bn_b)          (fc1)
// mode 3: out = res[m] + gelu(bn_g*BN_RS*(acc+bias)+bn_b) (fc2, final)
// ---------------------------------------------------------------------------
__global__ __launch_bounds__(128)
void gemm_bf16_kernel(const unsigned short* __restrict__ A,
                      const unsigned short* __restrict__ W,
                      const float* __restrict__ bias,
                      const float* __restrict__ bn_g,
                      const float* __restrict__ bn_b,
                      const float* __restrict__ res,
                      float* __restrict__ out,
                      int K, int N, int mode) {
  __shared__ __align__(16) unsigned short lA[2][64 * 32];
  __shared__ __align__(16) unsigned short lB[2][64 * 32];
  int bm = blockIdx.x * 64;
  int bn = blockIdx.y * 64;
  int tid  = threadIdx.x;
  int wave = tid >> 5, lane = tid & 31;
  int wm = (wave >> 1) * 32, wn = (wave & 1) * 32;
  int idx = lane & 15, half = lane >> 4;

  // per-thread staging chunks (16B each): c0 = tid, c1 = tid + 128
  int r0 = tid >> 2,         o0 = (tid & 3) * 8;
  int r1 = (tid + 128) >> 2, o1 = ((tid + 128) & 3) * 8;
  const unsigned short* gA0 = A + (size_t)(bm + r0) * K + o0;  // advance by 32/step
  const unsigned short* gA1 = A + (size_t)(bm + r1) * K + o1;
  const unsigned short* gB0 = W + (size_t)(bn + r0) * K + o0;
  const unsigned short* gB1 = W + (size_t)(bn + r1) * K + o1;
  int l0 = r0 * 32 + o0, l1 = r1 * 32 + o1;

  v8f acc[2][2];
  for (int i = 0; i < 2; i++)
    for (int j = 0; j < 2; j++)
      for (int r = 0; r < 8; r++) acc[i][j][r] = 0.f;

  int nsteps = K >> 5;

  auto issue = [&](int s, int buf) {
    const unsigned short* a0 = gA0 + s * 32;
    const unsigned short* a1 = gA1 + s * 32;
    const unsigned short* b0 = gB0 + s * 32;
    const unsigned short* b1 = gB1 + s * 32;
#if USE_ASYNC_LDS
    __builtin_amdgcn_global_load_async_to_lds_b128(
        (g_v4i*)(size_t)a0, (l_v4i*)&lA[buf][l0], 0, 0);
    __builtin_amdgcn_global_load_async_to_lds_b128(
        (g_v4i*)(size_t)a1, (l_v4i*)&lA[buf][l1], 0, 0);
    __builtin_amdgcn_global_load_async_to_lds_b128(
        (g_v4i*)(size_t)b0, (l_v4i*)&lB[buf][l0], 0, 0);
    __builtin_amdgcn_global_load_async_to_lds_b128(
        (g_v4i*)(size_t)b1, (l_v4i*)&lB[buf][l1], 0, 0);
#else
    *(uint4*)&lA[buf][l0] = *(const uint4*)a0;
    *(uint4*)&lA[buf][l1] = *(const uint4*)a1;
    *(uint4*)&lB[buf][l0] = *(const uint4*)b0;
    *(uint4*)&lB[buf][l1] = *(const uint4*)b1;
#endif
  };

  issue(0, 0);
  for (int i = 0; i < nsteps; i++) {
    int cur = i & 1;
#if USE_ASYNC_LDS
    __builtin_amdgcn_s_wait_asynccnt(0);
#endif
    __syncthreads();                  // buffer `cur` ready on all waves; also
                                      // proves reads of buffer cur^1 finished
    if (i + 1 < nsteps) issue(i + 1, cur ^ 1);  // overlaps with WMMAs below

    FragU af[2], bf[2];
    for (int ii = 0; ii < 2; ii++) {
      int row = wm + ii * 16 + idx;
      af[ii].q[0] = *(const uint4*)&lA[cur][row * 32 + half * 8];
      af[ii].q[1] = *(const uint4*)&lA[cur][row * 32 + half * 8 + 16];
    }
    for (int jj = 0; jj < 2; jj++) {
      int row = wn + jj * 16 + idx;
      bf[jj].q[0] = *(const uint4*)&lB[cur][row * 32 + half * 8];
      bf[jj].q[1] = *(const uint4*)&lB[cur][row * 32 + half * 8 + 16];
    }
    for (int ii = 0; ii < 2; ii++)
      for (int jj = 0; jj < 2; jj++)
        acc[ii][jj] = wmma_bf16(af[ii], bf[jj], acc[ii][jj]);
  }

  for (int i = 0; i < 2; i++)
    for (int j = 0; j < 2; j++) {
      int col = bn + wn + j * 16 + idx;
      float bc = bias[col];
      float sc = (mode >= 2) ? bn_g[col] * BN_RS : 1.f;
      float sh = (mode >= 2) ? bn_b[col] : 0.f;
      for (int r = 0; r < 8; r++) {
        int rowg = bm + wm + i * 16 + r + half * 8;
        float v = acc[i][j][r] + bc;
        if (mode == 0) {
          out[(size_t)rowg * N + col] = v;
        } else if (mode == 1) {
          int dst = swin_map(rowg);
          out[(size_t)dst * N + col] = v + res[(size_t)dst * N + col];
        } else if (mode == 2) {
          out[(size_t)rowg * N + col] = gelu_exact(sc * v + sh);
        } else {
          out[(size_t)rowg * N + col] =
              res[(size_t)rowg * N + col] + gelu_exact(sc * v + sh);
        }
      }
    }
}

// ---------------------------------------------------------------------------
// Attention: one wave per (window, head). N=64 tokens, hd=32.
// S = (q*scale) k^T  (16 wmma, K=32) + rel-pos bias + shift mask, softmax,
// O = P v (16 wmma, K=64 via LDS-staged P and transposed V).
// ---------------------------------------------------------------------------
__global__ __launch_bounds__(32)
void attn_kernel(const float* __restrict__ Y,      // [65536,1152] q|k|v
                 const float* __restrict__ rpb,    // [225,12]
                 unsigned short* __restrict__ out) // [65536,384] bf16 windowed
{
  __shared__ __align__(16) unsigned short Pl[64 * 64]; // softmax probs bf16
  __shared__ __align__(16) unsigned short Vt[32 * 64]; // V transposed [d][tok]
  int w = blockIdx.x, head = blockIdx.y;
  int lane = threadIdx.x;
  int idx = lane & 15, half = lane >> 4;
  int m0 = w * 64;
  int wh = (w >> 3) & 7, ww = w & 7;
  const float* Q  = Y + (size_t)m0 * QKV_N + head * 32;
  const float* Kp = Q + C_DIM;
  const float* Vp = Q + 2 * C_DIM;

  // Q (A frags, 4 row tiles) and K (B frags, 4 col tiles), f32 -> bf16
  FragU qf[4], kf[4];
  const float qscale = 0.17677669529663687f; // 32^-0.5
  for (int t = 0; t < 4; t++) {
    int row = t * 16 + idx;
    const float* pq = Q  + (size_t)row * QKV_N + half * 8;
    const float* pk = Kp + (size_t)row * QKV_N + half * 8;
    for (int e = 0; e < 8; e++) {
      qf[t].s[e]     = f2bf_u(pq[e]      * qscale);
      qf[t].s[8 + e] = f2bf_u(pq[16 + e] * qscale);
      kf[t].s[e]     = f2bf_u(pk[e]);
      kf[t].s[8 + e] = f2bf_u(pk[16 + e]);
    }
  }
  // stage V transposed into LDS
  for (int r = lane; r < 64; r += 32) {
    const float* pv = Vp + (size_t)r * QKV_N;
    for (int d = 0; d < 32; d++) Vt[d * 64 + r] = f2bf_u(pv[d]);
  }

  // scores + softmax, one 16-row strip at a time
  for (int rt = 0; rt < 4; rt++) {
    v8f sacc[4];
    for (int ct = 0; ct < 4; ct++) {
      for (int r = 0; r < 8; r++) sacc[ct][r] = 0.f;
      sacc[ct] = wmma_bf16(qf[rt], kf[ct], sacc[ct]);
    }
    float scs[4][8];
    for (int ct = 0; ct < 4; ct++)
      for (int r = 0; r < 8; r++) {
        int i = rt * 16 + r + half * 8;        // query token
        int j = ct * 16 + idx;                 // key token
        int ci = 15 * (i >> 3) + (i & 7);
        int jj = 63 - j;
        int cj = 15 * (jj >> 3) + (jj & 7);
        float bv = rpb[(ci + cj) * N_HEADS + head];
        int li = lab3(wh * 8 + (i >> 3)) * 3 + lab3(ww * 8 + (i & 7));
        int lj = lab3(wh * 8 + (j >> 3)) * 3 + lab3(ww * 8 + (j & 7));
        scs[ct][r] = sacc[ct][r] + bv + ((li != lj) ? -100.f : 0.f);
      }
    for (int r = 0; r < 8; r++) {
      float mx = fmaxf(fmaxf(scs[0][r], scs[1][r]), fmaxf(scs[2][r], scs[3][r]));
      for (int m = 8; m >= 1; m >>= 1) mx = fmaxf(mx, __shfl_xor(mx, m, 32));
      float sum = 0.f;
      for (int ct = 0; ct < 4; ct++) { scs[ct][r] = __expf(scs[ct][r] - mx); sum += scs[ct][r]; }
      for (int m = 8; m >= 1; m >>= 1) sum += __shfl_xor(sum, m, 32);
      float inv = 1.f / sum;
      int i = rt * 16 + r + half * 8;
      for (int ct = 0; ct < 4; ct++)
        Pl[i * 64 + ct * 16 + idx] = f2bf_u(scs[ct][r] * inv);
    }
  }
  __syncthreads();

  // O = P @ V : 4 row tiles x 2 col tiles, K = 64 (2 wmma each)
  v8f oacc[4][2];
  for (int rt = 0; rt < 4; rt++)
    for (int ct = 0; ct < 2; ct++)
      for (int r = 0; r < 8; r++) oacc[rt][ct][r] = 0.f;
  for (int rt = 0; rt < 4; rt++)
    for (int kt = 0; kt < 2; kt++) {
      FragU pf;
      int row = rt * 16 + idx;
      pf.q[0] = *(const uint4*)&Pl[row * 64 + kt * 32 + half * 8];
      pf.q[1] = *(const uint4*)&Pl[row * 64 + kt * 32 + half * 8 + 16];
      for (int ct = 0; ct < 2; ct++) {
        FragU vf;
        int d = ct * 16 + idx;
        vf.q[0] = *(const uint4*)&Vt[d * 64 + kt * 32 + half * 8];
        vf.q[1] = *(const uint4*)&Vt[d * 64 + kt * 32 + half * 8 + 16];
        oacc[rt][ct] = wmma_bf16(pf, vf, oacc[rt][ct]);
      }
    }
  for (int rt = 0; rt < 4; rt++)
    for (int ct = 0; ct < 2; ct++)
      for (int r = 0; r < 8; r++) {
        int tok = rt * 16 + r + half * 8;
        int d   = ct * 16 + idx;
        out[(size_t)(m0 + tok) * C_DIM + head * 32 + d] = f2bf_u(oacc[rt][ct][r]);
      }
}

// ---------------------------------------------------------------------------
// Depthwise 3x3 SAME + bias + BN + GELU, NHWC, 4 channels/thread, bf16 out.
// ---------------------------------------------------------------------------
__global__ __launch_bounds__(256)
void dwconv_kernel(const float* __restrict__ Z1, const float* __restrict__ dww,
                   const float* __restrict__ dwb, const float* __restrict__ bn_g,
                   const float* __restrict__ bn_b, unsigned short* __restrict__ Z2) {
  size_t t = (size_t)blockIdx.x * 256 + threadIdx.x;  // N_TOK * HID/4 threads
  int cg = (int)(t % (HID_DIM / 4));
  size_t pix = t / (HID_DIM / 4);
  int cid = cg * 4;
  int b = (int)(pix >> 12), y = (int)(pix >> 6) & 63, x = (int)pix & 63;
  float a[4] = {0.f, 0.f, 0.f, 0.f};
  for (int ky = -1; ky <= 1; ky++) {
    int yy = y + ky; if (yy < 0 || yy > 63) continue;
    for (int kx = -1; kx <= 1; kx++) {
      int xx = x + kx; if (xx < 0 || xx > 63) continue;
      const float* p = Z1 + (((size_t)b * 4096 + yy * 64 + xx) * HID_DIM + cid);
      float4 v = *(const float4*)p;
      int kidx = (ky + 1) * 3 + (kx + 1);
      a[0] += v.x * dww[(cid + 0) * 9 + kidx];
      a[1] += v.y * dww[(cid + 1) * 9 + kidx];
      a[2] += v.z * dww[(cid + 2) * 9 + kidx];
      a[3] += v.w * dww[(cid + 3) * 9 + kidx];
    }
  }
  unsigned short* o = Z2 + pix * HID_DIM + cid;
  for (int c = 0; c < 4; c++) {
    float v = a[c] + dwb[cid + c];
    v = gelu_exact(bn_g[cid + c] * BN_RS * v + bn_b[cid + c]);
    o[c] = f2bf_u(v);
  }
}

// ---------------------------------------------------------------------------
// Launch
// ---------------------------------------------------------------------------
extern "C" void kernel_launch(void* const* d_in, const int* in_sizes, int n_in,
                              void* d_out, int out_size, void* d_ws, size_t ws_size,
                              hipStream_t stream) {
  const float* x      = (const float*)d_in[0];
  const float* ln1_g  = (const float*)d_in[3];
  const float* ln1_b  = (const float*)d_in[4];
  const float* qkv_w  = (const float*)d_in[5];
  const float* qkv_b  = (const float*)d_in[6];
  const float* rpb    = (const float*)d_in[7];
  const float* proj_w = (const float*)d_in[8];
  const float* proj_b = (const float*)d_in[9];
  const float* ln2_g  = (const float*)d_in[10];
  const float* ln2_b  = (const float*)d_in[11];
  const float* fc1_w  = (const float*)d_in[12];
  const float* fc1_b  = (const float*)d_in[13];
  const float* bn1_g  = (const float*)d_in[14];
  const float* bn1_b  = (const float*)d_in[15];
  const float* dw_w   = (const float*)d_in[16];
  const float* dw_b   = (const float*)d_in[17];
  const float* bn2_g  = (const float*)d_in[18];
  const float* bn2_b  = (const float*)d_in[19];
  const float* fc2_w  = (const float*)d_in[20];
  const float* fc2_b  = (const float*)d_in[21];
  const float* bn3_g  = (const float*)d_in[22];
  const float* bn3_b  = (const float*)d_in[23];
  float* out = (float*)d_out;

  char* ws = (char*)d_ws;
  // bf16 weights (3.54 MB)
  unsigned short* wqkv  = (unsigned short*)(ws + 0);          // 1152x384
  unsigned short* wproj = (unsigned short*)(ws + 884736);     // 384x384
  unsigned short* wfc1  = (unsigned short*)(ws + 1179648);    // 1536x384
  unsigned short* wfc2  = (unsigned short*)(ws + 2359296);    // 384x1536
  // reused activation regions
  float*          R1f = (float*)(ws + 3538944);               // Yqkv(288MB)/Z1(384MB)
  unsigned short* R2h = (unsigned short*)(ws + 406192128ULL); // XW/H2 (48MB)
  unsigned short* R3h = (unsigned short*)(ws + 456523776ULL); // AttnOut/Z2 (192MB)
  float*          x1  = (float*)(ws + 657850368ULL);          // x+attn (96MB)

  f2bf_kernel<<<(442368 + 255) / 256, 256, 0, stream>>>(qkv_w,  wqkv,  442368);
  f2bf_kernel<<<(147456 + 255) / 256, 256, 0, stream>>>(proj_w, wproj, 147456);
  f2bf_kernel<<<(589824 + 255) / 256, 256, 0, stream>>>(fc1_w,  wfc1,  589824);
  f2bf_kernel<<<(589824 + 255) / 256, 256, 0, stream>>>(fc2_w,  wfc2,  589824);

  // LN1 + roll + window partition -> bf16
  ln_kernel<<<N_TOK / 8, 256, 0, stream>>>(x, ln1_g, ln1_b, R2h, 1);
  // QKV GEMM
  gemm_bf16_kernel<<<dim3(N_TOK / 64, QKV_N / 64), 128, 0, stream>>>(
      R2h, wqkv, qkv_b, nullptr, nullptr, nullptr, R1f, C_DIM, QKV_N, 0);
  // windowed MHSA
  attn_kernel<<<dim3(1024, N_HEADS), 32, 0, stream>>>(R1f, rpb, R3h);
  // proj GEMM + un-window/un-roll scatter + residual -> x1
  gemm_bf16_kernel<<<dim3(N_TOK / 64, C_DIM / 64), 128, 0, stream>>>(
      R3h, wproj, proj_b, nullptr, nullptr, x, x1, C_DIM, C_DIM, 1);
  // LN2
  ln_kernel<<<N_TOK / 8, 256, 0, stream>>>(x1, ln2_g, ln2_b, R2h, 0);
  // fc1 GEMM + BN1 + GELU
  gemm_bf16_kernel<<<dim3(N_TOK / 64, HID_DIM / 64), 128, 0, stream>>>(
      R2h, wfc1, fc1_b, bn1_g, bn1_b, nullptr, R1f, C_DIM, HID_DIM, 2);
  // depthwise 3x3 + BN2 + GELU -> bf16
  dwconv_kernel<<<(unsigned)((size_t)N_TOK * (HID_DIM / 4) / 256), 256, 0, stream>>>(
      R1f, dw_w, dw_b, bn2_g, bn2_b, R3h);
  // fc2 GEMM + BN3 + GELU + residual -> d_out
  gemm_bf16_kernel<<<dim3(N_TOK / 64, C_DIM / 64), 128, 0, stream>>>(
      R3h, wfc2, fc2_b, bn3_g, bn3_b, x1, out, HID_DIM, C_DIM, 3);

  (void)in_sizes; (void)n_in; (void)out_size; (void)ws_size;
}
